// Pointnet2Backbone_42064909697455
// MI455X (gfx1250) — compile-verified, matrix operand order checked by reading, not run on hardware
//
#include <hip/hip_runtime.h>
#include <hip/hip_bf16.h>
#include <stdint.h>

typedef __attribute__((ext_vector_type(16))) _Float16 v16h;
typedef __attribute__((ext_vector_type(8)))  float    v8f;

#define BATCH   64
#define NPTS    16384
#define NP1     32
#define NS1     32
#define NP2     16
#define NS2     32
#define NCLS    607
#define R1      0.2f
#define R2      0.4f

// ---------------------------------------------------------------------------
// WMMA helpers: D = A(16x32 f16) * B(32x16 f16) + C(16x16 f32), wave32.
// A: lane&15 = row M, lane>>4 = half; lane holds K = k0+h*8+{0..7} and
//    k0+16+h*8+{0..7}.  B mirrored with lane&15 = output column N.
// C/D: lane holds 8 values, M = h*8+r, N = lane&15.
// ---------------------------------------------------------------------------
__device__ __forceinline__ void frag_load(v16h& v, const _Float16* p) {
#pragma unroll
    for (int j = 0; j < 8; ++j) v[j] = p[j];
#pragma unroll
    for (int j = 0; j < 8; ++j) v[8 + j] = p[16 + j];
}

// One fused layer: out = relu(in * W^T + bias), fp16 LDS activations.
// in:  [rows][cin_p] fp16 (cin_p multiple of 32, zero padded)
// W:   [cout][cin_p] fp16 (BN scale folded, zero padded)  bias: [cout] fp32
// out: [rows][cout] fp16
__device__ __forceinline__ void wmma_layer(
    const _Float16* __restrict__ W, const float* __restrict__ bias,
    const _Float16* in, _Float16* out,
    int rows, int cin_p, int cout, int wave, int nwaves, int lane)
{
    const int col_tiles = cout >> 4;
    const int ntiles    = (rows >> 4) * col_tiles;
    const int half      = lane >> 4;
    const int l15       = lane & 15;
    for (int t = wave; t < ntiles; t += nwaves) {
        const int rt = t / col_tiles;
        const int ct = t - rt * col_tiles;
        const int n  = (ct << 4) + l15;
        const _Float16* arow = in + (size_t)((rt << 4) + l15) * cin_p + (half << 3);
        const _Float16* brow = W  + (size_t)n * cin_p + (half << 3);
        __builtin_prefetch(brow, 0, 0);
        v8f acc = {};
        for (int k0 = 0; k0 < cin_p; k0 += 32) {
            v16h a, b;
            frag_load(a, arow + k0);
            frag_load(b, brow + k0);
            acc = __builtin_amdgcn_wmma_f32_16x16x32_f16(
                false, a, false, b, (short)0, acc, false, false);
        }
        const float bn = bias[n];
#pragma unroll
        for (int r = 0; r < 8; ++r) {
            float v = acc[r] + bn;
            v = v > 0.f ? v : 0.f;
            out[(size_t)((rt << 4) + (half << 3) + r) * cout + n] = (_Float16)v;
        }
    }
}

// Final layer of each SA module: fuse relu + max over the `rows` (=nsample)
// axis into the epilogue.  Post-relu values are >=0 so int-punned atomicMax
// on an LDS float buffer (pre-initialized to 0) is exact.
__device__ __forceinline__ void wmma_layer_max(
    const _Float16* __restrict__ W, const float* __restrict__ bias,
    const _Float16* in, float* mxbuf,
    int rows, int cin_p, int cout, int wave, int nwaves, int lane)
{
    const int col_tiles = cout >> 4;
    const int ntiles    = (rows >> 4) * col_tiles;
    const int half      = lane >> 4;
    const int l15       = lane & 15;
    for (int t = wave; t < ntiles; t += nwaves) {
        const int rt = t / col_tiles;
        const int ct = t - rt * col_tiles;
        const int n  = (ct << 4) + l15;
        const _Float16* arow = in + (size_t)((rt << 4) + l15) * cin_p + (half << 3);
        const _Float16* brow = W  + (size_t)n * cin_p + (half << 3);
        v8f acc = {};
        for (int k0 = 0; k0 < cin_p; k0 += 32) {
            v16h a, b;
            frag_load(a, arow + k0);
            frag_load(b, brow + k0);
            acc = __builtin_amdgcn_wmma_f32_16x16x32_f16(
                false, a, false, b, (short)0, acc, false, false);
        }
        const float bn = bias[n];
        float m = 0.f;
#pragma unroll
        for (int r = 0; r < 8; ++r) {
            float v = acc[r] + bn;
            v = v > 0.f ? v : 0.f;
            m = fmaxf(m, v);
        }
        m = fmaxf(m, __shfl_xor(m, 16, 32));   // combine the two row-halves
        if (half == 0) atomicMax((int*)&mxbuf[n], __float_as_int(m));
    }
}

// ---------------------------------------------------------------------------
// Weight prep: fold BN into conv weights.  W'[o,c] = w[o,c]*g[o]*rsqrt(rv+eps)
// (zero padded to cin_p), b'[o] = b[o] - rm[o]*scale[o].
// ---------------------------------------------------------------------------
__global__ void prep_kernel(
    const float* __restrict__ w, const float* __restrict__ g,
    const float* __restrict__ bb, const float* __restrict__ rm,
    const float* __restrict__ rv,
    _Float16* __restrict__ Wq, float* __restrict__ Bq,
    int cout, int cin, int cinp)
{
    int i = blockIdx.x * 256 + threadIdx.x;
    if (i < cout) {
        float sc = g[i] * rsqrtf(rv[i] + 1e-5f);
        Bq[i] = bb[i] - rm[i] * sc;
    }
    int total = cout * cinp;
    if (i < total) {
        int o = i / cinp, c = i - o * cinp;
        float sc = g[o] * rsqrtf(rv[o] + 1e-5f);
        Wq[i] = (c < cin) ? (_Float16)(w[o * cin + c] * sc) : (_Float16)0.f;
    }
}

// ---------------------------------------------------------------------------
// FPS stage 1: one 256-thread block per batch, running-min distance array in
// global scratch, block-wide argmax with lowest-index tiebreak (jnp.argmax).
// ---------------------------------------------------------------------------
__global__ __launch_bounds__(256) void fps1_kernel(
    const float* __restrict__ xyz, float* __restrict__ dist,
    int* __restrict__ fidx, float* __restrict__ nxyz)
{
    const int b = blockIdx.x, tid = threadIdx.x;
    const float* x = xyz + (size_t)b * NPTS * 3;
    float* d = dist + (size_t)b * NPTS;
    __shared__ float rv_[256];
    __shared__ int   ri_[256];
    __shared__ float pc[3];
    for (int i = tid; i < NPTS; i += 256) d[i] = 1e10f;
    if (tid == 0) {
        fidx[b * NP1] = 0;
        pc[0] = x[0]; pc[1] = x[1]; pc[2] = x[2];
        nxyz[(size_t)b * NP1 * 3 + 0] = x[0];
        nxyz[(size_t)b * NP1 * 3 + 1] = x[1];
        nxyz[(size_t)b * NP1 * 3 + 2] = x[2];
    }
    __syncthreads();
    for (int s = 1; s < NP1; ++s) {
        float cx = pc[0], cy = pc[1], cz = pc[2];
        float bv = -1.f; int bi = 0;
        for (int i = tid; i < NPTS; i += 256) {
            float dx = x[3 * i] - cx, dy = x[3 * i + 1] - cy, dz = x[3 * i + 2] - cz;
            float nd = fminf(d[i], dx * dx + dy * dy + dz * dz);
            d[i] = nd;
            if (nd > bv) { bv = nd; bi = i; }
        }
        rv_[tid] = bv; ri_[tid] = bi;
        __syncthreads();
        for (int off = 128; off > 0; off >>= 1) {
            if (tid < off) {
                float ov = rv_[tid + off]; int oi = ri_[tid + off];
                if (ov > rv_[tid] || (ov == rv_[tid] && oi < ri_[tid])) {
                    rv_[tid] = ov; ri_[tid] = oi;
                }
            }
            __syncthreads();
        }
        if (tid == 0) {
            int ni = ri_[0];
            fidx[b * NP1 + s] = ni;
            float px = x[3 * ni], py = x[3 * ni + 1], pz = x[3 * ni + 2];
            pc[0] = px; pc[1] = py; pc[2] = pz;
            nxyz[((size_t)b * NP1 + s) * 3 + 0] = px;
            nxyz[((size_t)b * NP1 + s) * 3 + 1] = py;
            nxyz[((size_t)b * NP1 + s) * 3 + 2] = pz;
        }
        __syncthreads();
    }
}

// FPS stage 2: 32 source points -> one wave32 per batch, all-register.
__global__ __launch_bounds__(32) void fps2_kernel(
    const float* __restrict__ xyz1, int* __restrict__ fidx,
    float* __restrict__ nxyz)
{
    int b = blockIdx.x, lane = threadIdx.x;
    float px = xyz1[((size_t)b * NP1 + lane) * 3 + 0];
    float py = xyz1[((size_t)b * NP1 + lane) * 3 + 1];
    float pz = xyz1[((size_t)b * NP1 + lane) * 3 + 2];
    float dist = 1e10f;
    float cx = __shfl(px, 0, 32), cy = __shfl(py, 0, 32), cz = __shfl(pz, 0, 32);
    if (lane == 0) {
        fidx[b * NP2] = 0;
        nxyz[(size_t)b * NP2 * 3 + 0] = cx;
        nxyz[(size_t)b * NP2 * 3 + 1] = cy;
        nxyz[(size_t)b * NP2 * 3 + 2] = cz;
    }
    for (int s = 1; s < NP2; ++s) {
        float dx = px - cx, dy = py - cy, dz = pz - cz;
        dist = fminf(dist, dx * dx + dy * dy + dz * dz);
        float v = dist; int idx = lane;
        for (int off = 16; off > 0; off >>= 1) {
            float ov = __shfl_xor(v, off, 32);
            int   oi = __shfl_xor(idx, off, 32);
            if (ov > v || (ov == v && oi < idx)) { v = ov; idx = oi; }
        }
        cx = __shfl(px, idx, 32); cy = __shfl(py, idx, 32); cz = __shfl(pz, idx, 32);
        if (lane == 0) {
            fidx[b * NP2 + s] = idx;
            nxyz[((size_t)b * NP2 + s) * 3 + 0] = cx;
            nxyz[((size_t)b * NP2 + s) * 3 + 1] = cy;
            nxyz[((size_t)b * NP2 + s) * 3 + 2] = cz;
        }
    }
}

// ---------------------------------------------------------------------------
// Ball query + group for sa1: one wave per (b,s).  Collect the first NS1
// in-radius indices in index order via ballot/popc compaction; pad with the
// first hit (or 0).  Emit normalized fp16 rows padded K:3->32.
// ---------------------------------------------------------------------------
__global__ __launch_bounds__(256) void group1_kernel(
    const float* __restrict__ xyz, const float* __restrict__ nxyz1,
    _Float16* __restrict__ g1)
{
    __shared__ int lists[8][32];
    int lane = threadIdx.x & 31, wv = threadIdx.x >> 5;
    int w = blockIdx.x * 8 + wv;           // 0 .. B*NP1-1
    int b = w >> 5;
    const float* x = xyz + (size_t)b * NPTS * 3;
    float cx = nxyz1[(size_t)w * 3 + 0];
    float cy = nxyz1[(size_t)w * 3 + 1];
    float cz = nxyz1[(size_t)w * 3 + 2];
    int* list = lists[wv];
    int count = 0, first = -1;
    for (int base = 0; base < NPTS && count < NS1; base += 32) {
        int i = base + lane;
        float dx = x[3 * i] - cx, dy = x[3 * i + 1] - cy, dz = x[3 * i + 2] - cz;
        bool in = (dx * dx + dy * dy + dz * dz) < (R1 * R1);
        unsigned m32 = (unsigned)__ballot(in);
        if (first < 0 && m32) first = base + (__ffs(m32) - 1);
        int rank = __popc(m32 & ((1u << lane) - 1u));
        if (in && count + rank < NS1) list[count + rank] = i;
        count += __popc(m32);
    }
    if (first < 0) first = 0;
    if (count > NS1) count = NS1;
    if (lane >= count) list[lane] = first;
    __syncthreads();
    int idx = list[lane];
    _Float16* row = g1 + ((size_t)w * NS1 + lane) * 32;
    row[0] = (_Float16)((x[3 * idx + 0] - cx) * (1.f / R1));
    row[1] = (_Float16)((x[3 * idx + 1] - cy) * (1.f / R1));
    row[2] = (_Float16)((x[3 * idx + 2] - cz) * (1.f / R1));
#pragma unroll
    for (int c = 3; c < 32; ++c) row[c] = (_Float16)0.f;
}

// Ball query + group for sa2: 32 candidates fit one wave chunk exactly.
// Row = [3 normalized xyz | 128 f1 feats | pad to 160].
__global__ __launch_bounds__(256) void group2_kernel(
    const float* __restrict__ xyz1, const float* __restrict__ nxyz2,
    const _Float16* __restrict__ f1h, _Float16* __restrict__ g2)
{
    __shared__ int lists[8][32];
    int lane = threadIdx.x & 31, wv = threadIdx.x >> 5;
    int w = blockIdx.x * 8 + wv;           // 0 .. B*NP2-1
    int b = w >> 4;
    const float* x = xyz1 + (size_t)b * NP1 * 3;
    float cx = nxyz2[(size_t)w * 3 + 0];
    float cy = nxyz2[(size_t)w * 3 + 1];
    float cz = nxyz2[(size_t)w * 3 + 2];
    int* list = lists[wv];
    {
        int i = lane;
        float dx = x[3 * i] - cx, dy = x[3 * i + 1] - cy, dz = x[3 * i + 2] - cz;
        bool in = (dx * dx + dy * dy + dz * dz) < (R2 * R2);
        unsigned m32 = (unsigned)__ballot(in);
        int first = m32 ? (__ffs(m32) - 1) : 0;
        int rank = __popc(m32 & ((1u << lane) - 1u));
        int count = __popc(m32);
        if (in) list[rank] = i;
        if (lane >= count) list[lane] = first;
    }
    __syncthreads();
    for (int r = 0; r < NS2; ++r) {
        int idx = list[r];
        _Float16* row = g2 + ((size_t)w * NS2 + r) * 160;
        const _Float16* fr = f1h + ((size_t)b * NP1 + idx) * 128;
        for (int c = lane; c < 160; c += 32) {
            float v;
            if (c < 3) {
                float cc = (c == 0) ? cx : (c == 1) ? cy : cz;
                v = (x[3 * idx + c] - cc) * (1.f / R2);
            } else if (c < 131) {
                v = (float)fr[c - 3];
            } else {
                v = 0.f;
            }
            row[c] = (_Float16)v;
        }
    }
}

// sa3 grouping: concat [xyz2 | f2], pad 259 -> 288.
__global__ __launch_bounds__(256) void group3_kernel(
    const float* __restrict__ nxyz2, const _Float16* __restrict__ f2h,
    _Float16* __restrict__ g3)
{
    int i = blockIdx.x * 256 + threadIdx.x;
    if (i >= BATCH * NP2 * 288) return;
    int c = i % 288;
    int bk = i / 288;
    float v;
    if (c < 3)        v = nxyz2[(size_t)bk * 3 + c];
    else if (c < 259) v = (float)f2h[(size_t)bk * 256 + (c - 3)];
    else              v = 0.f;
    g3[i] = (_Float16)v;
}

// ---------------------------------------------------------------------------
// Fused WMMA MLP stacks (one workgroup = one group; activations in LDS).
// ---------------------------------------------------------------------------
__global__ __launch_bounds__(256) void mlp1_kernel(
    const _Float16* __restrict__ g1,
    const _Float16* __restrict__ W0, const float* __restrict__ B0,
    const _Float16* __restrict__ W1, const float* __restrict__ B1,
    const _Float16* __restrict__ W2, const float* __restrict__ B2,
    _Float16* __restrict__ f1h)
{
    __shared__ _Float16 x0[32 * 32];
    __shared__ _Float16 x1[32 * 64];
    __shared__ _Float16 x2[32 * 64];
    __shared__ float    mx[128];
    int tid = threadIdx.x, lane = tid & 31, wv = tid >> 5;
    const _Float16* src = g1 + (size_t)blockIdx.x * (32 * 32);
    for (int i = tid; i < 32 * 32; i += 256) x0[i] = src[i];
    for (int i = tid; i < 128; i += 256) mx[i] = 0.f;
    __syncthreads();
    wmma_layer(W0, B0, x0, x1, 32, 32, 64, wv, 8, lane);
    __syncthreads();
    wmma_layer(W1, B1, x1, x2, 32, 64, 64, wv, 8, lane);
    __syncthreads();
    wmma_layer_max(W2, B2, x2, mx, 32, 64, 128, wv, 8, lane);
    __syncthreads();
    for (int i = tid; i < 128; i += 256)
        f1h[(size_t)blockIdx.x * 128 + i] = (_Float16)mx[i];
}

__global__ __launch_bounds__(256) void mlp2_kernel(
    const _Float16* __restrict__ g2,
    const _Float16* __restrict__ W0, const float* __restrict__ B0,
    const _Float16* __restrict__ W1, const float* __restrict__ B1,
    const _Float16* __restrict__ W2, const float* __restrict__ B2,
    _Float16* __restrict__ f2h)
{
    __shared__ _Float16 x0[32 * 160];
    __shared__ _Float16 x1[32 * 128];
    __shared__ _Float16 x2[32 * 128];
    __shared__ float    mx[256];
    int tid = threadIdx.x, lane = tid & 31, wv = tid >> 5;
    const _Float16* src = g2 + (size_t)blockIdx.x * (32 * 160);
    for (int i = tid; i < 32 * 160; i += 256) x0[i] = src[i];
    for (int i = tid; i < 256; i += 256) mx[i] = 0.f;
    __syncthreads();
    wmma_layer(W0, B0, x0, x1, 32, 160, 128, wv, 8, lane);
    __syncthreads();
    wmma_layer(W1, B1, x1, x2, 32, 128, 128, wv, 8, lane);
    __syncthreads();
    wmma_layer_max(W2, B2, x2, mx, 32, 128, 256, wv, 8, lane);
    __syncthreads();
    for (int i = tid; i < 256; i += 256)
        f2h[(size_t)blockIdx.x * 256 + i] = (_Float16)mx[i];
}

__global__ __launch_bounds__(256) void mlp3_kernel(
    const _Float16* __restrict__ g3,
    const _Float16* __restrict__ W0, const float* __restrict__ B0,
    const _Float16* __restrict__ W1, const float* __restrict__ B1,
    const _Float16* __restrict__ W2, const float* __restrict__ B2,
    float* __restrict__ h)
{
    __shared__ _Float16 x0[16 * 288];
    __shared__ _Float16 x1[16 * 256];
    __shared__ _Float16 x2[16 * 512];
    __shared__ float    mx[128];
    int tid = threadIdx.x, lane = tid & 31, wv = tid >> 5;
    const _Float16* src = g3 + (size_t)blockIdx.x * (16 * 288);
    for (int i = tid; i < 16 * 288; i += 256) x0[i] = src[i];
    for (int i = tid; i < 128; i += 256) mx[i] = 0.f;
    __syncthreads();
    wmma_layer(W0, B0, x0, x1, 16, 288, 256, wv, 8, lane);
    __syncthreads();
    wmma_layer(W1, B1, x1, x2, 16, 256, 512, wv, 8, lane);
    __syncthreads();
    wmma_layer_max(W2, B2, x2, mx, 16, 512, 128, wv, 8, lane);
    __syncthreads();
    for (int i = tid; i < 128; i += 256)
        h[(size_t)blockIdx.x * 128 + i] = mx[i];
}

// Final FC 128 -> 607 in fp32 (tiny; one thread per output element).
__global__ __launch_bounds__(256) void fc_kernel(
    const float* __restrict__ h, const float* __restrict__ w,
    const float* __restrict__ bias, float* __restrict__ out)
{
    int i = blockIdx.x * 256 + threadIdx.x;
    if (i >= BATCH * NCLS) return;
    int b = i / NCLS, o = i - b * NCLS;
    const float* hb = h + b * 128;
    const float* wr = w + o * 128;
    float s = bias[o];
#pragma unroll 4
    for (int c = 0; c < 128; ++c) s += hb[c] * wr[c];
    out[i] = s;
}

// ---------------------------------------------------------------------------
extern "C" void kernel_launch(void* const* d_in, const int* in_sizes, int n_in,
                              void* d_out, int out_size, void* d_ws, size_t ws_size,
                              hipStream_t stream)
{
    (void)in_sizes; (void)n_in; (void)out_size; (void)ws_size;
    const float* pc  = (const float*)d_in[0];
    auto lw = [&](int L, int j) -> const float* { return (const float*)d_in[1 + L * 5 + j]; };
    const float* fcw = (const float*)d_in[46];
    const float* fcb = (const float*)d_in[47];

    static const int cins[9]  = {3, 64, 64, 131, 128, 128, 259, 256, 512};
    static const int couts[9] = {64, 64, 128, 128, 128, 256, 256, 512, 128};
    static const int cinps[9] = {32, 64, 64, 160, 128, 128, 288, 256, 512};

    char* ws = (char*)d_ws;
    size_t off = 0;
    auto alloc = [&](size_t bytes) -> char* {
        char* p = ws + off;
        off = (off + bytes + 255) & ~(size_t)255;
        return p;
    };

    _Float16* Wq[9]; float* Bq[9];
    for (int L = 0; L < 9; ++L) {
        Wq[L] = (_Float16*)alloc((size_t)couts[L] * cinps[L] * 2);
        Bq[L] = (float*)alloc((size_t)couts[L] * 4);
    }
    float*    dist  = (float*)alloc((size_t)BATCH * NPTS * 4);
    int*      fidx1 = (int*)alloc((size_t)BATCH * NP1 * 4);
    float*    nx1   = (float*)alloc((size_t)BATCH * NP1 * 3 * 4);
    _Float16* g1    = (_Float16*)alloc((size_t)BATCH * NP1 * NS1 * 32 * 2);
    _Float16* f1h   = (_Float16*)alloc((size_t)BATCH * NP1 * 128 * 2);
    int*      fidx2 = (int*)alloc((size_t)BATCH * NP2 * 4);
    float*    nx2   = (float*)alloc((size_t)BATCH * NP2 * 3 * 4);
    _Float16* g2    = (_Float16*)alloc((size_t)BATCH * NP2 * NS2 * 160 * 2);
    _Float16* f2h   = (_Float16*)alloc((size_t)BATCH * NP2 * 256 * 2);
    _Float16* g3    = (_Float16*)alloc((size_t)BATCH * NP2 * 288 * 2);
    float*    h3    = (float*)alloc((size_t)BATCH * 128 * 4);

    for (int L = 0; L < 9; ++L) {
        int total = couts[L] * cinps[L];
        int nth = total > couts[L] ? total : couts[L];
        prep_kernel<<<(nth + 255) / 256, 256, 0, stream>>>(
            lw(L, 0), lw(L, 1), lw(L, 2), lw(L, 3), lw(L, 4),
            Wq[L], Bq[L], couts[L], cins[L], cinps[L]);
    }
    fps1_kernel<<<BATCH, 256, 0, stream>>>(pc, dist, fidx1, nx1);
    group1_kernel<<<(BATCH * NP1) / 8, 256, 0, stream>>>(pc, nx1, g1);
    mlp1_kernel<<<BATCH * NP1, 256, 0, stream>>>(
        g1, Wq[0], Bq[0], Wq[1], Bq[1], Wq[2], Bq[2], f1h);
    fps2_kernel<<<BATCH, 32, 0, stream>>>(nx1, fidx2, nx2);
    group2_kernel<<<(BATCH * NP2) / 8, 256, 0, stream>>>(nx1, nx2, f1h, g2);
    mlp2_kernel<<<BATCH * NP2, 256, 0, stream>>>(
        g2, Wq[3], Bq[3], Wq[4], Bq[4], Wq[5], Bq[5], f2h);
    group3_kernel<<<(BATCH * NP2 * 288 + 255) / 256, 256, 0, stream>>>(nx2, f2h, g3);
    mlp3_kernel<<<BATCH, 256, 0, stream>>>(
        g3, Wq[6], Bq[6], Wq[7], Bq[7], Wq[8], Bq[8], h3);
    fc_kernel<<<(BATCH * NCLS + 255) / 256, 256, 0, stream>>>(h3, fcw, fcb, (float*)d_out);
}
